// GaussianResampling_51737176048313
// MI455X (gfx1250) — matrix operand order repaired and checked by reading, not compiled, not run on hardware
//
#include <hip/hip_runtime.h>

// GaussianResampling: out(B,T,D) = softmax_s(-(t - c_s)^2 / 10) @ x(B,S,D)
// B=16, S=512, D=768, T=4096, variance=10.
// Strategy: per block = one (batch, 16-frame) tile.
//   1) LDS scan of durations -> token centers (mask = d>0, sentinel center otherwise)
//   2) unnormalized exp weights -> f16 in LDS, per-frame sums in f32
//   3) W(16x512) @ X(512x768) via v_wmma_f32_16x16x32_f16, X staged f32->f16
//      transposed in LDS in 32-deep K chunks (two 384-wide D halves, fully
//      unrolled so the 6 v8f accumulators stay in VGPRs -- no scratch spills)
//   4) scale accumulators by 1/sum(frame) and store f32

typedef __attribute__((ext_vector_type(16))) _Float16 v16h;
typedef __attribute__((ext_vector_type(8)))  _Float16 v8h;
typedef __attribute__((ext_vector_type(8)))  float    v8f;

#define NB 16
#define NS 512
#define ND 768
#define NT 4096
#define INV_VAR 0.1f

#define WSTRIDE 520   // halves: W row stride (16B aligned, de-conflicted)
#define XSTRIDE 40    // halves: xt column stride (16B aligned, de-conflicted)
#define HALF_D 384

__global__ __launch_bounds__(256) void gauss_resample_wmma(
    const float* __restrict__ x,   // (B,S,D) f32
    const float* __restrict__ dur, // (B,S)   f32
    float* __restrict__ out)       // (B,T,D) f32
{
  __shared__ __attribute__((aligned(16))) _Float16 wle[16 * WSTRIDE];
  __shared__ __attribute__((aligned(16))) _Float16 xt[HALF_D * XSTRIDE];
  __shared__ float sc[NS];
  __shared__ float cc[NS];
  __shared__ float fsum[16];

  const int tid  = threadIdx.x;
  const int lane = tid & 31;
  const int wv   = tid >> 5;          // wave 0..7
  const int b    = blockIdx.y;
  const int tb   = blockIdx.x * 16;   // frame tile base (0-based)

  // ---- 1. inclusive scan of durations -> token end times ----
  const float dd0 = dur[b * NS + tid];
  const float dd1 = dur[b * NS + tid + 256];
  sc[tid]       = dd0;
  sc[tid + 256] = dd1;
  if (tid < 16) fsum[tid] = 0.0f;
  __syncthreads();
  for (int off = 1; off < NS; off <<= 1) {
    float v0 = (tid >= off) ? sc[tid - off] : 0.0f;
    float v1 = sc[tid + 256 - off];          // always in range (off <= 256)
    __syncthreads();
    sc[tid]       += v0;
    sc[tid + 256] += v1;
    __syncthreads();
  }
  // centers; masked tokens (d==0) -> huge sentinel so exp underflows to 0
  cc[tid]       = (dd0 > 0.0f) ? (sc[tid]       - 0.5f * dd0) : 3.0e18f;
  cc[tid + 256] = (dd1 > 0.0f) ? (sc[tid + 256] - 0.5f * dd1) : 3.0e18f;
  __syncthreads();

  // ---- 2. unnormalized Gaussian weights (f16) + per-frame denominators ----
  {
    const int f = tid >> 4;            // frame within tile 0..15
    const int j = tid & 15;
    const float tval = (float)(tb + f + 1);   // t = 1..T
    float ps = 0.0f;
    for (int s = j; s < NS; s += 16) {
      float diff = tval - cc[s];
      float e = __expf(-(diff * diff) * INV_VAR);  // max logit <= 0: safe
      ps += e;
      wle[f * WSTRIDE + s] = (_Float16)e;
    }
    atomicAdd(&fsum[f], ps);
  }

  // ---- 3. GEMM: 16x768 tile = W(16x512) x X[b](512x768) via f16 WMMA ----
  v8f acc[6] = {};                     // 6 D-tiles of 16 per wave (registers)

  const int n    = lane & 15;          // B/C/D column within tile
  const int hi   = lane >> 4;          // lane half
  const int kb   = hi * 8;             // A-frag K base within chunk
  const int soff = hi * 8;             // B-frag K base within chunk
  const int m    = n;                  // A-frag row (frame)

#pragma unroll                          // <- half must be compile-time so the
  for (int half = 0; half < 2; ++half) //    acc[] indices are constants (no spill)
  {
    const int dbase = half * HALF_D;
    for (int kt = 0; kt < NS / 32; ++kt) {
      __syncthreads();  // protect xt / (first iter: wle) from prior readers
      // stage X chunk [32 x 384] f32 -> f16, transposed into LDS.
      // 32*96 float4 / 256 threads = exactly 12 uniform iterations.
      const float4* src =
          (const float4*)(x + ((size_t)b * NS + kt * 32) * ND + dbase);
      for (int it = 0; it < 12; ++it) {
        int i  = it * 256 + tid;
        int s  = i / 96;                 // row 0..31 within chunk
        int dc = (i - s * 96) * 4;       // col 0..380 within half
        float4 v = src[s * (ND / 4) + (dc >> 2)];
        xt[(dc + 0) * XSTRIDE + s] = (_Float16)v.x;
        xt[(dc + 1) * XSTRIDE + s] = (_Float16)v.y;
        xt[(dc + 2) * XSTRIDE + s] = (_Float16)v.z;
        xt[(dc + 3) * XSTRIDE + s] = (_Float16)v.w;
      }
      __syncthreads();

      // A fragment (ISA 16-bit 16x32 layout): two contiguous b128 LDS reads
      const _Float16* wrow = &wle[m * WSTRIDE + kt * 32 + kb];
      v8h alo = *(const v8h*)(wrow);
      v8h ahi = *(const v8h*)(wrow + 16);
      v16h a = __builtin_shufflevector(alo, ahi,
          0,1,2,3,4,5,6,7,8,9,10,11,12,13,14,15);

#pragma unroll
      for (int q3 = 0; q3 < 3; ++q3) {
        const int q  = half * 3 + q3;            // constant after unroll
        const int dt = wv + 8 * q;               // global D tile 0..47
        const int nl = dt * 16 - dbase + n;      // column in staged slice
        const _Float16* xcol = &xt[nl * XSTRIDE + soff];
        v8h blo = *(const v8h*)(xcol);
        v8h bhi = *(const v8h*)(xcol + 16);
        v16h bb = __builtin_shufflevector(blo, bhi,
            0,1,2,3,4,5,6,7,8,9,10,11,12,13,14,15);
        acc[q] = __builtin_amdgcn_wmma_f32_16x16x32_f16(
            /*neg_a=*/false, a, /*neg_b=*/false, bb,
            /*c_mod=*/(short)0, acc[q], /*reuse_a=*/false, /*reuse_b=*/false);
      }
    }
  }
  __syncthreads();

  // ---- 4. fold softmax denominator into f32 accumulators, store ----
  const int mh = hi * 8;               // C/D layout: VGPR r -> M = r + 8*hi
  float inv[8];
#pragma unroll
  for (int r = 0; r < 8; ++r) inv[r] = 1.0f / fsum[mh + r];

#pragma unroll
  for (int q = 0; q < 6; ++q) {
    const int dcol = (wv + 8 * q) * 16 + n;
    float* op = out + ((size_t)b * NT + tb + mh) * ND + dcol;
#pragma unroll
    for (int r = 0; r < 8; ++r)
      op[(size_t)r * ND] = acc[q][r] * inv[r];
  }
}

extern "C" void kernel_launch(void* const* d_in, const int* in_sizes, int n_in,
                              void* d_out, int out_size, void* d_ws, size_t ws_size,
                              hipStream_t stream) {
  (void)in_sizes; (void)n_in; (void)d_ws; (void)ws_size; (void)out_size;
  const float* x   = (const float*)d_in[0];  // (B,S,D)
  const float* dur = (const float*)d_in[1];  // (B,S)
  // d_in[2] (mask) unused: recovered as dur > 0. d_in[3] (T) is fixed = 4096.
  float* out = (float*)d_out;                // (B,T,D)
  dim3 grid(NT / 16, NB);
  gauss_resample_wmma<<<grid, 256, 0, stream>>>(x, dur, out);
}
